// EncoderLayer_76261439308167
// MI455X (gfx1250) — matrix operand order repaired and checked by reading, not compiled, not run on hardware
//
#include <hip/hip_runtime.h>

typedef _Float16 half_t;
typedef __attribute__((ext_vector_type(8)))  _Float16 v8h;
typedef __attribute__((ext_vector_type(16))) _Float16 v16h;
typedef __attribute__((ext_vector_type(8)))  float    v8f;
typedef __attribute__((ext_vector_type(4)))  int      v4i;

union V16 { v16h v; v8h h[2]; };

#define D_MODEL 1024
#define D_FF    4096
#define NHEAD   16
#define DK      64
#define BATCH   2
#define SEQ     2048
#define NROWS   (BATCH*SEQ)   /* 4096 */

#ifndef __has_builtin
#define __has_builtin(x) 0
#endif

#if defined(__AMDGCN__) && __has_builtin(__builtin_amdgcn_global_load_async_to_lds_b128)
#define HAVE_ASYNC 1
#else
#define HAVE_ASYNC 0
#endif

#if HAVE_ASYNC
__device__ __forceinline__ void async_b128(const half_t* g, half_t* l) {
    __builtin_amdgcn_global_load_async_to_lds_b128(
        (v4i*)g,
        (__attribute__((address_space(3))) v4i*)l,
        0, 0);
}
#if __has_builtin(__builtin_amdgcn_s_wait_asynccnt)
#define WAIT_ASYNC(n) __builtin_amdgcn_s_wait_asynccnt(n)
#else
#define WAIT_ASYNC(n) asm volatile("s_wait_asynccnt %0" :: "n"(n) : "memory")
#endif
#endif

// ---------------------------------------------------------------------------
// f32 -> f16 conversion
// ---------------------------------------------------------------------------
__global__ __launch_bounds__(256)
void cvt_f32_f16(const float* __restrict__ s, half_t* __restrict__ d, int n) {
    int i = blockIdx.x * 256 + threadIdx.x;
    if (i < n) d[i] = (half_t)s[i];
}

// ---------------------------------------------------------------------------
// f32 [K][N] -> f16 [N][K] transposed convert (LDS 32x32 tile, coalesced)
// ---------------------------------------------------------------------------
__global__ __launch_bounds__(256)
void cvt_transpose_f16(const float* __restrict__ src, half_t* __restrict__ dst,
                       int K, int N) {
    __shared__ float tile[32][33];
    const int tx = threadIdx.x & 31;
    const int ty = threadIdx.x >> 5;        // 0..7
    const int n0 = blockIdx.x * 32;
    const int k0 = blockIdx.y * 32;
    for (int i = 0; i < 4; ++i)
        tile[ty + 8 * i][tx] = src[(size_t)(k0 + ty + 8 * i) * N + n0 + tx];
    __syncthreads();
    for (int i = 0; i < 4; ++i)
        dst[(size_t)(n0 + ty + 8 * i) * K + k0 + tx] =
            (half_t)tile[tx][ty + 8 * i];
}

// ---------------------------------------------------------------------------
// Tiled WMMA GEMM: C[M,N] = A[M,K] @ Bt[N,K]^T + bias, optional relu.
// Block tile 128x128, K-step 64, double-buffered async LDS staging.
// 256 threads = 8 waves in 4x2, wave tile 32x64.
// Dynamic LDS: 2 bufs x (A 128x72 + B 128x72) halves = 73728 bytes.
// ---------------------------------------------------------------------------
#define GEMM_LDS_BYTES (2 * 2 * 128 * 72 * 2)

__global__ __launch_bounds__(256)
void gemm_wmma(const half_t* __restrict__ A,   // [M][K]
               const half_t* __restrict__ Bt,  // [N][K]
               const float* __restrict__ bias, float* __restrict__ Cf,
               half_t* __restrict__ Ch, int M, int N, int K, int relu)
{
    extern __shared__ char smem_raw[];
    half_t* const As = (half_t*)smem_raw;                 // [2][128*72]
    half_t* const Bs = (half_t*)smem_raw + 2 * 128 * 72;  // [2][128*72]

    const int t    = threadIdx.x;
    const int lane = t & 31;
    const int wave = t >> 5;
    const int wr   = wave >> 1;   // 0..3
    const int wc   = wave & 1;    // 0..1
    const int m0   = blockIdx.y * 128;
    const int n0   = blockIdx.x * 128;
    const int ln   = lane & 15;
    const int kbA  = (lane < 16) ? 0 : 16;   // A-frag K base
    const int kbB  = (lane < 16) ? 0 : 16;   // B-frag K base
    (void)M;

    v8f acc[2][4];
    for (int i = 0; i < 2; ++i)
        for (int j = 0; j < 4; ++j)
            for (int r = 0; r < 8; ++r) acc[i][j][r] = 0.f;

    const int knum = K >> 6;

    auto issue = [&](int kt, int buf) {
        const int k0 = kt << 6;
        half_t* const ab = As + buf * (128 * 72);
        half_t* const bb = Bs + buf * (128 * 72);
        for (int p = 0; p < 4; ++p) {
            const int c   = t + p * 256;          // 1024 chunks of 8 halves
            const int row = c >> 3;
            const int c8  = (c & 7) << 3;
            const half_t* ga = &A [(size_t)(m0 + row) * K + k0 + c8];
            const half_t* gb = &Bt[(size_t)(n0 + row) * K + k0 + c8];
            half_t* la = &ab[row * 72 + c8];
            half_t* lb = &bb[row * 72 + c8];
#if HAVE_ASYNC
            async_b128(ga, la);
            async_b128(gb, lb);
#else
            *(v8h*)la = *(const v8h*)ga;
            *(v8h*)lb = *(const v8h*)gb;
#endif
        }
    };

    issue(0, 0);
    if (knum > 1) issue(1, 1);

    for (int kt = 0; kt < knum; ++kt) {
        const int cur = kt & 1;
#if HAVE_ASYNC
        if (kt + 1 < knum) WAIT_ASYNC(8);   // this tile's 8 asyncs done
        else               WAIT_ASYNC(0);
#endif
        __syncthreads();
        const half_t* ab = As + cur * (128 * 72);
        const half_t* bb = Bs + cur * (128 * 72);
        for (int ks = 0; ks < 2; ++ks) {
            V16 a[2];
            for (int mf = 0; mf < 2; ++mf) {
                const half_t* p = &ab[(32 * wr + 16 * mf + ln) * 72 + 32 * ks + kbA];
                a[mf].h[0] = *(const v8h*)p;
                a[mf].h[1] = *(const v8h*)(p + 8);
            }
            for (int nf = 0; nf < 4; ++nf) {
                V16 b;
                const half_t* p = &bb[(64 * wc + 16 * nf + ln) * 72 + 32 * ks + kbB];
                b.h[0] = *(const v8h*)p;
                b.h[1] = *(const v8h*)(p + 8);
                for (int mf = 0; mf < 2; ++mf)
                    acc[mf][nf] = __builtin_amdgcn_wmma_f32_16x16x32_f16(
                        false, a[mf].v, false, b.v, (short)0, acc[mf][nf],
                        false, false);
            }
        }
        __syncthreads();                 // everyone done reading buf `cur`
        if (kt + 2 < knum) issue(kt + 2, cur);
    }

    const int roff = (lane < 16) ? 0 : 8;
    for (int nf = 0; nf < 4; ++nf) {
        const int n  = n0 + 64 * wc + 16 * nf + ln;
        const float bv = bias ? bias[n] : 0.f;
        for (int mf = 0; mf < 2; ++mf) {
            const int mbase = m0 + 32 * wr + 16 * mf + roff;
            for (int r = 0; r < 8; ++r) {
                float v = acc[mf][nf][r] + bv;
                if (relu) v = v > 0.f ? v : 0.f;
                const size_t idx = (size_t)(mbase + r) * N + n;
                if (Cf) Cf[idx] = v;
                if (Ch) Ch[idx] = (half_t)v;
            }
        }
    }
}

// ---------------------------------------------------------------------------
// Flash attention: one block per (q-tile of 64, head, batch). 4 waves,
// each wave owns 16 q rows. Online softmax, O accumulated in f32.
// ---------------------------------------------------------------------------
__global__ __launch_bounds__(128)
void attn_wmma(const half_t* __restrict__ Q, const half_t* __restrict__ Kh,
               const half_t* __restrict__ Vh, const int* __restrict__ mask,
               half_t* __restrict__ ctx)
{
    __shared__ half_t Ks[64 * 72];   // [key][d]
    __shared__ half_t Vt[64 * 72];   // [d][key]  (transposed V)
    __shared__ half_t Ps[64 * 72];   // [qrow][key] probabilities
    __shared__ float  msk[64];

    const int t    = threadIdx.x;
    const int lane = t & 31;
    const int w    = t >> 5;
    const int ln   = lane & 15;
    const int kbA  = (lane < 16) ? 0 : 16;
    const int kbB  = (lane < 16) ? 0 : 16;
    const int q0   = blockIdx.x * 64;
    const int h    = blockIdx.y;
    const int b    = blockIdx.z;
    const size_t rowbase = (size_t)b * SEQ;
    const int col0 = h * DK;

    // Q fragments for this wave's 16 rows (A-matrix 16x64 -> 2 frags)
    V16 qf[2];
    {
        const half_t* qp = &Q[(rowbase + q0 + 16 * w + ln) * D_MODEL + col0];
        for (int f = 0; f < 2; ++f) {
            qf[f].h[0] = *(const v8h*)(qp + 32 * f + kbA);
            qf[f].h[1] = *(const v8h*)(qp + 32 * f + kbA + 8);
        }
    }

    v8f  o[4];
    float Mr[8], Lr[8];
    for (int nt = 0; nt < 4; ++nt)
        for (int r = 0; r < 8; ++r) o[nt][r] = 0.f;
    for (int r = 0; r < 8; ++r) { Mr[r] = -3.0e38f; Lr[r] = 0.f; }

    for (int j = 0; j < SEQ / 64; ++j) {
        const int k0 = j * 64;
        __syncthreads();   // previous iteration's LDS reads complete
        // K tile: async DMA copy [key][d]; V tile: batched loads, transpose
        for (int p = 0; p < 4; ++p) {
            const int c   = t + p * 128;     // 512 chunks of 8 halves
            const int row = c >> 3;
            const int c8  = (c & 7) << 3;
            const half_t* gk = &Kh[(rowbase + k0 + row) * D_MODEL + col0 + c8];
#if HAVE_ASYNC
            async_b128(gk, &Ks[row * 72 + c8]);
#else
            *(v8h*)&Ks[row * 72 + c8] = *(const v8h*)gk;
#endif
        }
        v8h vv[4];
        int vrow[4], vc8[4];
        for (int p = 0; p < 4; ++p) {
            const int c = t + p * 128;
            vrow[p] = c >> 3;
            vc8[p]  = (c & 7) << 3;
            vv[p] = *(const v8h*)&Vh[(rowbase + k0 + vrow[p]) * D_MODEL + col0 + vc8[p]];
        }
        for (int p = 0; p < 4; ++p)
            for (int i = 0; i < 8; ++i)
                Vt[(vc8[p] + i) * 72 + vrow[p]] = vv[p][i];
        if (t < 64) msk[t] = (mask[b * SEQ + k0 + t] == 0) ? -1.0e10f : 0.f;
#if HAVE_ASYNC
        WAIT_ASYNC(0);
#endif
        __syncthreads();

        // scores S = Q x K^T  (contraction over d=64 -> 2 WMMA per n-tile)
        v8f sc[4];
        for (int nt = 0; nt < 4; ++nt) {
            V16 kb0, kb1;
            const half_t* kp = &Ks[(16 * nt + ln) * 72 + kbB];
            kb0.h[0] = *(const v8h*)kp;        kb0.h[1] = *(const v8h*)(kp + 8);
            kb1.h[0] = *(const v8h*)(kp + 32); kb1.h[1] = *(const v8h*)(kp + 40);
            v8f z; for (int r = 0; r < 8; ++r) z[r] = 0.f;
            z = __builtin_amdgcn_wmma_f32_16x16x32_f16(
                    false, qf[0].v, false, kb0.v, (short)0, z, false, false);
            sc[nt] = __builtin_amdgcn_wmma_f32_16x16x32_f16(
                    false, qf[1].v, false, kb1.v, (short)0, z, false, false);
        }

        // scale + mask + online softmax (row r in lanes 0-15, r+8 in 16-31)
        float mnew[8];
        for (int r = 0; r < 8; ++r) {
            float mx = -3.0e38f;
            for (int nt = 0; nt < 4; ++nt) {
                float s = sc[nt][r] * 0.125f + msk[16 * nt + ln];
                sc[nt][r] = s;
                mx = fmaxf(mx, s);
            }
            for (int off = 8; off >= 1; off >>= 1)
                mx = fmaxf(mx, __shfl_xor(mx, off, 32));
            mnew[r] = mx;
        }
        const int prow = 16 * w + ((lane < 16) ? 0 : 8);
        for (int r = 0; r < 8; ++r) {
            float M2    = fmaxf(Mr[r], mnew[r]);
            float alpha = __expf(Mr[r] - M2);
            Mr[r] = M2;
            float srow = 0.f;
            for (int nt = 0; nt < 4; ++nt) {
                float p = __expf(sc[nt][r] - M2);
                sc[nt][r] = p;
                srow += p;
            }
            for (int off = 8; off >= 1; off >>= 1)
                srow += __shfl_xor(srow, off, 32);
            Lr[r] = Lr[r] * alpha + srow;
            for (int nt = 0; nt < 4; ++nt) o[nt][r] *= alpha;
            for (int nt = 0; nt < 4; ++nt)
                Ps[(prow + r) * 72 + 16 * nt + ln] = (half_t)sc[nt][r];
        }
        __syncthreads();

        // O += P x V
        V16 pf[2];
        {
            const half_t* pp = &Ps[(16 * w + ln) * 72];
            for (int f = 0; f < 2; ++f) {
                pf[f].h[0] = *(const v8h*)(pp + 32 * f + kbA);
                pf[f].h[1] = *(const v8h*)(pp + 32 * f + kbA + 8);
            }
        }
        for (int nt = 0; nt < 4; ++nt) {
            V16 vb0, vb1;
            const half_t* vp = &Vt[(16 * nt + ln) * 72 + kbB];
            vb0.h[0] = *(const v8h*)vp;        vb0.h[1] = *(const v8h*)(vp + 8);
            vb1.h[0] = *(const v8h*)(vp + 32); vb1.h[1] = *(const v8h*)(vp + 40);
            o[nt] = __builtin_amdgcn_wmma_f32_16x16x32_f16(
                false, pf[0].v, false, vb0.v, (short)0, o[nt], false, false);
            o[nt] = __builtin_amdgcn_wmma_f32_16x16x32_f16(
                false, pf[1].v, false, vb1.v, (short)0, o[nt], false, false);
        }
    }

    const int roff = (lane < 16) ? 0 : 8;
    for (int r = 0; r < 8; ++r) {
        const float inv = 1.f / Lr[r];
        const size_t row = rowbase + q0 + 16 * w + r + roff;
        for (int nt = 0; nt < 4; ++nt)
            ctx[row * D_MODEL + col0 + 16 * nt + ln] = (half_t)(o[nt][r] * inv);
    }
}

// ---------------------------------------------------------------------------
// y = LayerNorm(X + Y) * g + beta ; one block per row of 1024
// ---------------------------------------------------------------------------
__global__ __launch_bounds__(256)
void add_ln(const float* __restrict__ X, const float* __restrict__ Y,
            const float* __restrict__ g, const float* __restrict__ be,
            float* __restrict__ outf, half_t* __restrict__ outh)
{
    __shared__ float red[256];
    const int row = blockIdx.x, t = threadIdx.x;
    const float* xp = X + (size_t)row * D_MODEL;
    const float* yp = Y + (size_t)row * D_MODEL;
    float v[4];
    float s = 0.f;
    for (int i = 0; i < 4; ++i) {
        v[i] = xp[t + 256 * i] + yp[t + 256 * i];
        s += v[i];
    }
    red[t] = s; __syncthreads();
    for (int o = 128; o > 0; o >>= 1) { if (t < o) red[t] += red[t + o]; __syncthreads(); }
    const float mu = red[0] * (1.f / D_MODEL);
    __syncthreads();
    s = 0.f;
    for (int i = 0; i < 4; ++i) { float d = v[i] - mu; s += d * d; }
    red[t] = s; __syncthreads();
    for (int o = 128; o > 0; o >>= 1) { if (t < o) red[t] += red[t + o]; __syncthreads(); }
    const float rstd = rsqrtf(red[0] * (1.f / D_MODEL) + 1e-5f);
    for (int i = 0; i < 4; ++i) {
        int c = t + 256 * i;
        float yv = (v[i] - mu) * rstd * g[c] + be[c];
        if (outf) outf[(size_t)row * D_MODEL + c] = yv;
        if (outh) outh[(size_t)row * D_MODEL + c] = (half_t)yv;
    }
}

// ---------------------------------------------------------------------------
extern "C" void kernel_launch(void* const* d_in, const int* in_sizes, int n_in,
                              void* d_out, int out_size, void* d_ws, size_t ws_size,
                              hipStream_t stream) {
    (void)in_sizes; (void)n_in; (void)out_size; (void)ws_size;
    const float* x     = (const float*)d_in[0];
    const int*   mask  = (const int*)  d_in[1];
    const float* Wq    = (const float*)d_in[2];
    const float* bq    = (const float*)d_in[3];
    const float* Wk    = (const float*)d_in[4];
    const float* bk    = (const float*)d_in[5];
    const float* Wv    = (const float*)d_in[6];
    const float* bv    = (const float*)d_in[7];
    const float* Wo    = (const float*)d_in[8];
    const float* bo    = (const float*)d_in[9];
    const float* W1    = (const float*)d_in[10];
    const float* b1    = (const float*)d_in[11];
    const float* W2    = (const float*)d_in[12];
    const float* b2    = (const float*)d_in[13];
    const float* ln1g  = (const float*)d_in[14];
    const float* ln1b  = (const float*)d_in[15];
    const float* ln2g  = (const float*)d_in[16];
    const float* ln2b  = (const float*)d_in[17];
    float* out = (float*)d_out;

    char* ws = (char*)d_ws;
    size_t off = 0;
    auto alloc = [&](size_t bytes) -> char* {
        char* p = ws + off;
        off += (bytes + 255) & ~(size_t)255;
        return p;
    };
    const size_t NE  = (size_t)NROWS * D_MODEL;      // 4M
    const size_t WE  = (size_t)D_MODEL * D_MODEL;    // 1M
    const size_t FE  = (size_t)D_MODEL * D_FF;       // 4M
    const size_t HE  = (size_t)NROWS * D_FF;         // 16M

    half_t* xh   = (half_t*)alloc(NE * 2);
    half_t* wqt  = (half_t*)alloc(WE * 2);   // transposed [N][K]
    half_t* wkt  = (half_t*)alloc(WE * 2);
    half_t* wvt  = (half_t*)alloc(WE * 2);
    half_t* wot  = (half_t*)alloc(WE * 2);
    half_t* w1t  = (half_t*)alloc(FE * 2);   // [D_FF][D_MODEL]
    half_t* w2t  = (half_t*)alloc(FE * 2);   // [D_MODEL][D_FF]
    half_t* qh   = (half_t*)alloc(NE * 2);
    half_t* kh   = (half_t*)alloc(NE * 2);
    half_t* vh   = (half_t*)alloc(NE * 2);
    half_t* ctxh = (half_t*)alloc(NE * 2);
    float*  ao   = (float*) alloc(NE * 4);
    float*  x1   = (float*) alloc(NE * 4);
    half_t* x1h  = (half_t*)alloc(NE * 2);
    half_t* h1h  = (half_t*)alloc(HE * 2);
    float*  ffo  = (float*) alloc(NE * 4);

    // convert activations; convert+transpose weights ([K][N] -> [N][K])
    cvt_f32_f16<<<(unsigned)((NE + 255) / 256), 256, 0, stream>>>(x, xh, (int)NE);
    auto cvtT = [&](const float* s, half_t* d, int K, int N) {
        dim3 g(N / 32, K / 32);
        cvt_transpose_f16<<<g, 256, 0, stream>>>(s, d, K, N);
    };
    cvtT(Wq, wqt, D_MODEL, D_MODEL);
    cvtT(Wk, wkt, D_MODEL, D_MODEL);
    cvtT(Wv, wvt, D_MODEL, D_MODEL);
    cvtT(Wo, wot, D_MODEL, D_MODEL);
    cvtT(W1, w1t, D_MODEL, D_FF);
    cvtT(W2, w2t, D_FF, D_MODEL);

    dim3 blk(256);
    dim3 g_dd(D_MODEL / 128, NROWS / 128);   // (8, 32)
    dim3 g_df(D_FF / 128,    NROWS / 128);   // (32, 32)

    // Q/K/V projections (f16 out)
    gemm_wmma<<<g_dd, blk, GEMM_LDS_BYTES, stream>>>(xh, wqt, bq, nullptr, qh, NROWS, D_MODEL, D_MODEL, 0);
    gemm_wmma<<<g_dd, blk, GEMM_LDS_BYTES, stream>>>(xh, wkt, bk, nullptr, kh, NROWS, D_MODEL, D_MODEL, 0);
    gemm_wmma<<<g_dd, blk, GEMM_LDS_BYTES, stream>>>(xh, wvt, bv, nullptr, vh, NROWS, D_MODEL, D_MODEL, 0);

    // flash attention
    dim3 g_attn(SEQ / 64, NHEAD, BATCH);
    attn_wmma<<<g_attn, dim3(128), 0, stream>>>(qh, kh, vh, mask, ctxh);

    // output projection (f32 out)
    gemm_wmma<<<g_dd, blk, GEMM_LDS_BYTES, stream>>>(ctxh, wot, bo, ao, nullptr, NROWS, D_MODEL, D_MODEL, 0);

    // residual + LN1 (f32 + f16 out)
    add_ln<<<NROWS, blk, 0, stream>>>(x, ao, ln1g, ln1b, x1, x1h);

    // FFN
    gemm_wmma<<<g_df, blk, GEMM_LDS_BYTES, stream>>>(x1h, w1t, b1, nullptr, h1h, NROWS, D_FF, D_MODEL, 1);
    gemm_wmma<<<g_dd, blk, GEMM_LDS_BYTES, stream>>>(h1h, w2t, b2, ffo, nullptr, NROWS, D_MODEL, D_FF, 0);

    // residual + LN2 -> final output (f32)
    add_ln<<<NROWS, blk, 0, stream>>>(x1, ffo, ln2g, ln2b, out, nullptr);
}
